// CDCRB_27943057228099
// MI455X (gfx1250) — matrix operand order repaired and checked by reading, not compiled, not run on hardware
//
#include <hip/hip_runtime.h>

// ---------------------------------------------------------------------------
// Problem constants (match setup_inputs)
// ---------------------------------------------------------------------------
#define BB 2
#define QQ 2048
#define HH 48
#define WW 48
#define CC 64
#define EE 20
#define NGRP (BB*QQ*9)     /* 36864 ensemble groups (b,q,q1q2)            */
#define NROWS (NGRP*4)     /* 147456 MLP rows (x4 vx,vy members)          */

// ---------------------------------------------------------------------------
// CDNA5 WMMA types
// ---------------------------------------------------------------------------
typedef __attribute__((ext_vector_type(16))) __bf16        v16bf;
typedef __attribute__((ext_vector_type(8)))  float         v8f;
typedef __attribute__((ext_vector_type(4)))  unsigned int  v4u;

union FragBF { v16bf v; v4u q[2]; };

static __device__ __forceinline__ unsigned short f2bf(float f) {
  unsigned u = __float_as_uint(f);
  u += 0x7FFFu + ((u >> 16) & 1u);      // round-to-nearest-even bf16
  return (unsigned short)(u >> 16);
}

// ---------------------------------------------------------------------------
// Workspace layout (bytes)
// ---------------------------------------------------------------------------
#define OFF_FEAT_A   0u
#define OFF_FEAT_T   2359296u
#define OFF_FEAT_B   4718592u
#define OFF_PAD      7077888u     /* bf16 (B,C,50,50)                      */
#define OFF_WPK      7718144u     /* packed bf16 imnet weights            */
#define OFF_META_IX  8430848u
#define OFF_META_IY  9020672u
#define OFF_REL0     9610496u
#define OFF_REL1     10200320u
#define OFF_WROW     10790144u
#define OFF_RETSP    11379968u    /* (B,Q,3,9) f32                        */
#define OFF_CPIN     11822336u    /* (B,Q,10) f32                         */
#define OFF_CPRED    11986176u    /* (B,Q,9) f32                          */
#define OFF_CMEAN    12133632u
#define OFF_FUSED    12133888u

// packed-weight sub-offsets (ushort elements)
#define WPK_L1 0
#define WPK_L2 155648
#define WPK_L3 221184
#define WPK_L4 286720
#define WPK_L5 352256

// ---------------------------------------------------------------------------
// 3x3 SAME conv, NCHW/OIHW, fp32 (encoder: 2.5 GFLOP total, VALU path)
// ---------------------------------------------------------------------------
__global__ void k_conv3x3(const float* __restrict__ in, const float* __restrict__ w,
                          const float* __restrict__ base, float* __restrict__ out,
                          int Cin, int relu_in)
{
  int idx = blockIdx.x * blockDim.x + threadIdx.x;
  if (idx >= BB*CC*HH*WW) return;
  int x = idx % WW; int t = idx / WW;
  int y = t % HH;   t /= HH;
  int co = t % CC;  int b = t / CC;
  float acc = 0.f;
  for (int ci = 0; ci < Cin; ci++) {
    const float* ip = in + ((size_t)(b*Cin + ci))*HH*WW;
    const float* wp = w + ((size_t)(co*Cin + ci))*9;
#pragma unroll
    for (int kh = 0; kh < 3; kh++) {
      int yy = y + kh - 1;
      if (yy < 0 || yy >= HH) continue;
#pragma unroll
      for (int kw = 0; kw < 3; kw++) {
        int xx = x + kw - 1;
        if (xx < 0 || xx >= WW) continue;
        float v = ip[yy*WW + xx];
        if (relu_in) v = fmaxf(v, 0.f);
        acc += v * wp[kh*3 + kw];
      }
    }
  }
  if (base) acc += base[idx];
  out[idx] = acc;
}

// zero-padded bf16 feature map (B,C,50,50) -> the unfold becomes pure gather
__global__ void k_padbf(const float* __restrict__ feat, unsigned short* __restrict__ dst)
{
  int idx = blockIdx.x * blockDim.x + threadIdx.x;
  if (idx >= BB*CC*50*50) return;
  int x = idx % 50; int t = idx / 50;
  int y = t % 50;   t /= 50;
  int c = t % CC;   int b = t / CC;
  float v = 0.f;
  if (y >= 1 && y <= HH && x >= 1 && x <= WW)
    v = feat[((size_t)(b*CC + c)*HH + (y-1))*WW + (x-1)];
  dst[idx] = f2bf(v);
}

// ---------------------------------------------------------------------------
// Pack fp32 (K,N) weights into the exact WMMA B-fragment lane layout:
//   packed[((kc*NT+nt)*32 + lane)*16 + e] = W[kc*32 + (lane>>4)*8 + (e&7) + (e>>3)*16][nt*16 + (lane&15)]
// so a B fragment is two contiguous global_load_b128 per lane.
// ---------------------------------------------------------------------------
__global__ void k_packw(const float* __restrict__ W, unsigned short* __restrict__ dst,
                        int Kreal, int Nreal, int KC, int NT)
{
  int idx = blockIdx.x * blockDim.x + threadIdx.x;
  if (idx >= KC*NT*512) return;
  int e    = idx & 15;
  int lane = (idx >> 4) & 31;
  int tmp  = idx >> 9;
  int nt   = tmp % NT;
  int kc   = tmp / NT;
  int k = kc*32 + (lane >> 4)*8 + (e & 7) + (e >> 3)*16;
  int n = nt*16 + (lane & 15);
  float v = (k < Kreal && n < Nreal) ? W[(size_t)k*Nreal + n] : 0.f;
  dst[idx] = f2bf(v);
}

__global__ void k_cellmean(const float* __restrict__ cell, float* __restrict__ cm)
{
  int t = threadIdx.x;
  if (t < BB*2) {
    int b = t >> 1, d = t & 1;
    float s = 0.f;
    for (int q = 0; q < QQ; q++) s += cell[(size_t)(b*QQ + q)*2 + d];
    cm[t] = s * (1.f / (float)QQ);
  }
}

// Per ensemble group (b,q,q1,q2): gather indices, rel, swapped area weights.
__global__ void k_meta(const float* __restrict__ coord, const float* __restrict__ cell,
                       const float* __restrict__ cm,
                       int* __restrict__ ixA, int* __restrict__ iyA,
                       float* __restrict__ r0A, float* __restrict__ r1A,
                       float* __restrict__ wA, float* __restrict__ cpin)
{
  int gid = blockIdx.x * blockDim.x + threadIdx.x;
  if (gid >= NGRP) return;
  int bq = gid / 9, g = gid - bq*9;
  int b  = bq >> 11;
  int q1 = g/3 - 1, q2 = g%3 - 1;
  float cm0 = cm[b*2+0], cm1 = cm[b*2+1];
  float co0 = coord[(size_t)bq*2+0], co1 = coord[(size_t)bq*2+1];
  const float rx = 1.f/HH, ry = 1.f/WW;
  float area[4], rl0[4], rl1[4];
#pragma unroll
  for (int j = 0; j < 4; j++) {
    float vx = (j < 2) ? -1.f : 1.f;
    float vy = (j & 1) ?  1.f : -1.f;
    float s0 = vx*rx + (float)q1*cm0 + 1e-6f;
    float s1 = vy*ry + (float)q2*cm1 + 1e-6f;
    float c0 = fminf(fmaxf(co0 + s0, -1.f + 1e-6f), 1.f - 1e-6f);
    float c1 = fminf(fmaxf(co1 + s1, -1.f + 1e-6f), 1.f - 1e-6f);
    float fy = fminf(fmaxf(rintf((c0 + 1.f)*0.5f*(HH-1)), 0.f), (float)(HH-1));
    float fx = fminf(fmaxf(rintf((c1 + 1.f)*0.5f*(WW-1)), 0.f), (float)(WW-1));
    int iy = (int)fy, ix = (int)fx;
    float qc0 = -1.f + (2.f*iy + 1.f)/HH;
    float qc1 = -1.f + (2.f*ix + 1.f)/WW;
    float r0 = (co0 - qc0)*HH, r1 = (co1 - qc1)*WW;
    area[j] = fabsf(r0*r1) + 1e-9f;
    rl0[j] = r0; rl1[j] = r1;
    int row = gid*4 + j;
    ixA[row] = ix; iyA[row] = iy; r0A[row] = r0; r1A[row] = r1;
  }
  float tot = area[0] + area[1] + area[2] + area[3];
#pragma unroll
  for (int j = 0; j < 4; j++) wA[gid*4 + j] = area[3-j] / tot;   // local-ensemble swap
  if (g == 4) {
    float* cp = cpin + (size_t)bq*10;
#pragma unroll
    for (int j = 0; j < 4; j++) { cp[2*j] = rl0[j]; cp[2*j+1] = rl1[j]; }
    cp[8] = cell[(size_t)bq*2+0]*HH;
    cp[9] = cell[(size_t)bq*2+1]*WW;
  }
}

// ---------------------------------------------------------------------------
// Fused gather + 5-layer MLP on WMMA.
// 32 rows / WG (8 complete area-groups).  8 waves, each wave: 1 M-tile x 4 N-tiles.
// KC is a compile-time constant so the K-loop fully unrolls: straight-line
// WMMA code, no loop-carried accumulator rotation, loads hoist over WMMAs.
// ---------------------------------------------------------------------------
template<int KC, int SSTRIDE>
static __device__ __forceinline__ void mlp_layer256(
    const unsigned short* __restrict__ src,
    unsigned short* __restrict__ dst,
    const unsigned short* __restrict__ wp, const float* __restrict__ bias,
    int lane, int wave)
{
  const int mt     = wave >> 2;
  const int ntBase = (wave & 3) * 4;
  const int khalf  = lane >> 4;
  const int arow   = mt*16 + (lane & 15);
  v8f acc[4];
  const v8f z = {0.f,0.f,0.f,0.f,0.f,0.f,0.f,0.f};
#pragma unroll
  for (int i = 0; i < 4; i++) acc[i] = z;
  const v4u* wq = (const v4u*)wp;
  const int abase = arow*SSTRIDE + khalf*8;
  const int bbase = (ntBase*32 + lane) * 2;
#pragma unroll
  for (int kc = 0; kc < KC; kc++) {
    // A fragment: two 16B LDS loads per lane
    FragBF a;
    a.q[0] = *(const v4u*)(&src[abase + kc*32]);
    a.q[1] = *(const v4u*)(&src[abase + kc*32 + 16]);
    // 4 B fragments: eight contiguous 16B global loads, issued together
    FragBF bf[4];
#pragma unroll
    for (int i = 0; i < 4; i++) {
      int bidx = bbase + kc*1024 + i*64;   // ((kc*16+nt)*32+lane)*2
      bf[i].q[0] = wq[bidx];
      bf[i].q[1] = wq[bidx + 1];
    }
    // 4 independent back-to-back WMMAs
#pragma unroll
    for (int i = 0; i < 4; i++)
      acc[i] = __builtin_amdgcn_wmma_f32_16x16x32_bf16(
                 false, a.v, false, bf[i].v, (short)0, acc[i], false, false);
  }
  const int col0 = lane & 15;
#pragma unroll
  for (int i = 0; i < 4; i++) {
    int col = (ntBase + i)*16 + col0;
    float bv = bias[col];
#pragma unroll
    for (int rr = 0; rr < 8; rr++) {
      int row = mt*16 + khalf*8 + rr;
      float v = acc[i][rr] + bv;
      v = fmaxf(v, 0.f);
      dst[row*256 + col] = f2bf(v);
    }
  }
}

__global__ __launch_bounds__(256) void k_mlp(
    const unsigned short* __restrict__ wpk,
    const float* __restrict__ b1, const float* __restrict__ b2,
    const float* __restrict__ b3, const float* __restrict__ b4,
    const float* __restrict__ b5,
    const unsigned short* __restrict__ featpad,
    const float* __restrict__ cell,
    const int* __restrict__ ixA, const int* __restrict__ iyA,
    const float* __restrict__ r0A, const float* __restrict__ r1A,
    const float* __restrict__ wA,
    float* __restrict__ retsp)
{
  __shared__ __align__(16) unsigned short Xs[32*608];  // X / H2 / H4 (ping)
  __shared__ __align__(16) unsigned short Hs[32*256];  // H1 / H3     (pong)
  __shared__ float Ds[32*16];
  __shared__ float wr[32];
  const int tid = threadIdx.x;
  const int r0  = blockIdx.x * 32;

  // ---- gather: X[row] = [unfold9(feat) | rel | rel_cell | zero-pad to 608]
  {
    int rl = tid >> 3, sub = tid & 7;
    int r  = r0 + rl;
    int gid = r >> 2;
    int bq  = gid / 9;
    int b = bq >> 11, q = bq & (QQ-1);
    int iy = iyA[r], ix = ixA[r];
    const unsigned short* fp = featpad + (size_t)b*CC*2500;
    unsigned short* xr = &Xs[rl*608];
    for (int i = 0; i < 8; i++) {
      int c = sub*8 + i;
      const unsigned short* fc = fp + (size_t)c*2500 + iy*50 + ix;
#pragma unroll
      for (int kh = 0; kh < 3; kh++)
#pragma unroll
        for (int kw = 0; kw < 3; kw++)
          xr[c*9 + kh*3 + kw] = fc[kh*50 + kw];
    }
    if (sub == 0) {
      xr[576] = f2bf(r0A[r]);
      xr[577] = f2bf(r1A[r]);
      xr[578] = f2bf(cell[(size_t)(b*QQ + q)*2 + 0]*HH);
      xr[579] = f2bf(cell[(size_t)(b*QQ + q)*2 + 1]*WW);
      for (int k = 580; k < 608; k++) xr[k] = 0;
      wr[rl] = wA[r];
    }
  }
  __syncthreads();

  const int lane = tid & 31, wave = tid >> 5;
  mlp_layer256<19, 608>(Xs, Hs, wpk + WPK_L1, b1, lane, wave);
  __syncthreads();
  mlp_layer256< 8, 256>(Hs, Xs, wpk + WPK_L2, b2, lane, wave);
  __syncthreads();
  mlp_layer256< 8, 256>(Xs, Hs, wpk + WPK_L3, b3, lane, wave);
  __syncthreads();
  mlp_layer256< 8, 256>(Hs, Xs, wpk + WPK_L4, b4, lane, wave);
  __syncthreads();

  // ---- layer 5: N padded 3->16, one N-tile, waves 0..1
  if (wave < 2) {
    const v4u* wq = (const v4u*)(wpk + WPK_L5);
    int khalf = lane >> 4;
    int arow  = wave*16 + (lane & 15);
    int abase = arow*256 + khalf*8;
    v8f acc = {0.f,0.f,0.f,0.f,0.f,0.f,0.f,0.f};
#pragma unroll
    for (int kc = 0; kc < 8; kc++) {
      FragBF a;
      a.q[0] = *(const v4u*)(&Xs[abase + kc*32]);
      a.q[1] = *(const v4u*)(&Xs[abase + kc*32 + 16]);
      FragBF bfr;
      int bidx = (kc*32 + lane) * 2;
      bfr.q[0] = wq[bidx];
      bfr.q[1] = wq[bidx + 1];
      acc = __builtin_amdgcn_wmma_f32_16x16x32_bf16(
              false, a.v, false, bfr.v, (short)0, acc, false, false);
    }
    int col = lane & 15;
    float bv = (col < 3) ? b5[col] : 0.f;
#pragma unroll
    for (int rr = 0; rr < 8; rr++) {
      int row = wave*16 + khalf*8 + rr;
      Ds[row*16 + col] = acc[rr] + bv;
    }
  }
  __syncthreads();

  // ---- area-weighted local-ensemble combine (WG-local, no atomics)
  if (tid < 24) {
    int gl = tid / 3, c = tid % 3;
    float v = 0.f;
#pragma unroll
    for (int j = 0; j < 4; j++) v += Ds[(gl*4 + j)*16 + c] * wr[gl*4 + j];
    int gid = blockIdx.x*8 + gl;
    int bq = gid / 9, g = gid - bq*9;
    retsp[((size_t)bq*3 + c)*9 + g] = v;
  }
}

// ---------------------------------------------------------------------------
// Routing MLP + gaussian expert prior + expert-pool fusion (tiny)
// ---------------------------------------------------------------------------
__global__ void k_routing(const float* __restrict__ cell,
                          const float* __restrict__ W0, const float* __restrict__ b0,
                          const float* __restrict__ W1, const float* __restrict__ b1r,
                          const float* __restrict__ pool, float* __restrict__ fused)
{
  __shared__ float hbuf[BB*64];
  __shared__ float zS[BB*EE];
  __shared__ float rwS[BB*EE];
  int t = threadIdx.x;
  if (t < BB*64) {
    int b = t >> 6, n = t & 63;
    float rc0 = cell[(size_t)(b*QQ)*2 + 0]*HH;
    float rc1 = cell[(size_t)(b*QQ)*2 + 1]*WW;
    hbuf[t] = fmaxf(rc0*W0[n] + rc1*W0[64 + n] + b0[n], 0.f);
  }
  __syncthreads();
  if (t < BB*EE) {
    int b = t / EE, e = t - b*EE;
    float z = b1r[e];
    for (int n = 0; n < 64; n++) z += hbuf[b*64 + n]*W1[n*EE + e];
    zS[t] = z;
  }
  __syncthreads();
  if (t < BB) {
    int b = t;
    float mx = -1e30f;
    for (int e = 0; e < EE; e++) mx = fmaxf(mx, zS[b*EE + e]);
    float w0l[EE], s = 0.f;
    for (int e = 0; e < EE; e++) { w0l[e] = __expf(zS[b*EE + e] - mx); s += w0l[e]; }
    float inv = 1.f/s;
    float central = rintf(2.f / cell[(size_t)(b*QQ)*2 + 0]) / (float)WW;
    const float coef = 0.7978845608f;            // 1/(sqrt(2*pi)*0.5)
    float gl[EE], mg = -1e30f;
    for (int e = 0; e < EE; e++) {
      float d = 1.f + (float)e*(998.f/19.f) - central;
      gl[e] = coef*__expf(-2.f*d*d);
      mg = fmaxf(mg, gl[e]);
    }
    float sg = 0.f;
    for (int e = 0; e < EE; e++) { gl[e] = __expf(gl[e] - mg); sg += gl[e]; }
    float invg = 1.f/sg;
    for (int e = 0; e < EE; e++) rwS[b*EE + e] = w0l[e]*inv + gl[e]*invg;
  }
  __syncthreads();
  if (t < BB*81) {
    int b = t / 81, o = t - b*81;
    float v = 0.f;
    for (int e = 0; e < EE; e++) v += rwS[b*EE + e]*pool[e*81 + o];
    fused[t] = v;
  }
}

// per-query residual 3x3 conv with the per-batch fused expert kernel
__global__ void k_resconv(float* __restrict__ retsp, const float* __restrict__ fused)
{
  int bq = blockIdx.x*blockDim.x + threadIdx.x;
  if (bq >= BB*QQ) return;
  int b = bq >> 11;
  float* rp = retsp + (size_t)bq*27;
  float x[3][3][3];
#pragma unroll
  for (int ci = 0; ci < 3; ci++)
#pragma unroll
    for (int y = 0; y < 3; y++)
#pragma unroll
      for (int xx = 0; xx < 3; xx++)
        x[ci][y][xx] = rp[ci*9 + y*3 + xx];
  const float* fw = fused + b*81;
  float res[27];
#pragma unroll
  for (int o = 0; o < 3; o++)
#pragma unroll
    for (int y = 0; y < 3; y++)
#pragma unroll
      for (int xx = 0; xx < 3; xx++) {
        float acc = 0.f;
#pragma unroll
        for (int ci = 0; ci < 3; ci++)
#pragma unroll
          for (int kh = 0; kh < 3; kh++)
#pragma unroll
            for (int kw = 0; kw < 3; kw++) {
              int yy = y + kh - 1, x2 = xx + kw - 1;
              if (yy >= 0 && yy < 3 && x2 >= 0 && x2 < 3)
                acc += x[ci][yy][x2]*fw[((o*3 + ci)*3 + kh)*3 + kw];
            }
        res[o*9 + y*3 + xx] = acc;
      }
#pragma unroll
  for (int i = 0; i < 27; i++) rp[i] += res[i];
}

// imnet2: 10 -> 64 -> 64 -> 9, one block per query, LDS staged
__global__ __launch_bounds__(64) void k_imnet2(
    const float* __restrict__ cpin,
    const float* __restrict__ W0, const float* __restrict__ b0,
    const float* __restrict__ W1, const float* __restrict__ b1r,
    const float* __restrict__ W2, const float* __restrict__ b2r,
    float* __restrict__ cpred)
{
  __shared__ float xin[10];
  __shared__ float h1[64];
  __shared__ float h2[64];
  int bq = blockIdx.x;
  int t = threadIdx.x;
  if (t < 10) xin[t] = cpin[(size_t)bq*10 + t];
  __syncthreads();
  float a = b0[t];
  for (int i = 0; i < 10; i++) a += xin[i]*W0[i*64 + t];
  h1[t] = fmaxf(a, 0.f);
  __syncthreads();
  float a2 = b1r[t];
  for (int i = 0; i < 64; i++) a2 += h1[i]*W1[i*64 + t];
  h2[t] = fmaxf(a2, 0.f);
  __syncthreads();
  if (t < 9) {
    float a3 = b2r[t];
    for (int i = 0; i < 64; i++) a3 += h2[i]*W2[i*9 + t];
    cpred[(size_t)bq*9 + t] = a3;
  }
}

// out[b,q,c] = sum_k retsp[b,q,c,k] * conv_pred[b,q,k]
__global__ void k_final(const float* __restrict__ retsp, const float* __restrict__ cpred,
                        float* __restrict__ out)
{
  int idx = blockIdx.x*blockDim.x + threadIdx.x;
  if (idx >= BB*QQ*3) return;
  int c = idx % 3; int bq = idx / 3;
  float v = 0.f;
#pragma unroll
  for (int k = 0; k < 9; k++) v += retsp[((size_t)bq*3 + c)*9 + k]*cpred[(size_t)bq*9 + k];
  out[idx] = v;
}

// ---------------------------------------------------------------------------
extern "C" void kernel_launch(void* const* d_in, const int* in_sizes, int n_in,
                              void* d_out, int out_size, void* d_ws, size_t ws_size,
                              hipStream_t stream)
{
  (void)in_sizes; (void)n_in; (void)out_size; (void)ws_size;
  const float* inp    = (const float*)d_in[0];
  const float* coord  = (const float*)d_in[1];
  const float* cell   = (const float*)d_in[2];
  const float* w_head = (const float*)d_in[3];
  const float* wr1    = (const float*)d_in[4];
  const float* wr2    = (const float*)d_in[5];
  const float* pool   = (const float*)d_in[6];
  const float* rW0    = (const float*)d_in[7];
  const float* rb0    = (const float*)d_in[8];
  const float* rW1    = (const float*)d_in[9];
  const float* rb1    = (const float*)d_in[10];
  const float* iw[5]; const float* ib[5];
  for (int i = 0; i < 5; i++) { iw[i] = (const float*)d_in[11 + 2*i]; ib[i] = (const float*)d_in[12 + 2*i]; }
  const float* jw[3]; const float* jb[3];
  for (int i = 0; i < 3; i++) { jw[i] = (const float*)d_in[21 + 2*i]; jb[i] = (const float*)d_in[22 + 2*i]; }
  float* out = (float*)d_out;
  char* ws = (char*)d_ws;

  float*          featA   = (float*)(ws + OFF_FEAT_A);
  float*          featT   = (float*)(ws + OFF_FEAT_T);
  float*          featB   = (float*)(ws + OFF_FEAT_B);
  unsigned short* featpad = (unsigned short*)(ws + OFF_PAD);
  unsigned short* wpk     = (unsigned short*)(ws + OFF_WPK);
  int*            ixA     = (int*)(ws + OFF_META_IX);
  int*            iyA     = (int*)(ws + OFF_META_IY);
  float*          r0A     = (float*)(ws + OFF_REL0);
  float*          r1A     = (float*)(ws + OFF_REL1);
  float*          wA      = (float*)(ws + OFF_WROW);
  float*          retsp   = (float*)(ws + OFF_RETSP);
  float*          cpin    = (float*)(ws + OFF_CPIN);
  float*          cpred   = (float*)(ws + OFF_CPRED);
  float*          cmean   = (float*)(ws + OFF_CMEAN);
  float*          fusedW  = (float*)(ws + OFF_FUSED);

  // --- encoder: head + 4 resblocks (ping-pong) ---
  const int CONV_BLKS = (BB*CC*HH*WW + 255) / 256;
  k_conv3x3<<<CONV_BLKS, 256, 0, stream>>>(inp, w_head, nullptr, featA, 3, 0);
  float* cur = featA; float* nxt = featB;
  for (int i = 0; i < 4; i++) {
    k_conv3x3<<<CONV_BLKS, 256, 0, stream>>>(cur,  wr1 + (size_t)i*CC*CC*9, nullptr, featT, CC, 0);
    k_conv3x3<<<CONV_BLKS, 256, 0, stream>>>(featT, wr2 + (size_t)i*CC*CC*9, cur,    nxt,  CC, 1);
    float* t2 = cur; cur = nxt; nxt = t2;
  }
  k_padbf<<<(BB*CC*2500 + 255)/256, 256, 0, stream>>>(cur, featpad);

  // --- pack imnet weights into WMMA B-fragment layout (bf16) ---
  k_packw<<<(19*16*512 + 255)/256, 256, 0, stream>>>(iw[0], wpk + WPK_L1, 580, 256, 19, 16);
  k_packw<<<( 8*16*512 + 255)/256, 256, 0, stream>>>(iw[1], wpk + WPK_L2, 256, 256,  8, 16);
  k_packw<<<( 8*16*512 + 255)/256, 256, 0, stream>>>(iw[2], wpk + WPK_L3, 256, 256,  8, 16);
  k_packw<<<( 8*16*512 + 255)/256, 256, 0, stream>>>(iw[3], wpk + WPK_L4, 256, 256,  8, 16);
  k_packw<<<( 8* 1*512 + 255)/256, 256, 0, stream>>>(iw[4], wpk + WPK_L5, 256,   3,  8,  1);

  // --- per-row ensemble metadata ---
  k_cellmean<<<1, 64, 0, stream>>>(cell, cmean);
  k_meta<<<(NGRP + 255)/256, 256, 0, stream>>>(coord, cell, cmean, ixA, iyA, r0A, r1A, wA, cpin);

  // --- the big one: 147456-row WMMA MLP + local-ensemble combine ---
  k_mlp<<<NROWS/32, 256, 0, stream>>>(wpk, ib[0], ib[1], ib[2], ib[3], ib[4],
                                      featpad, cell, ixA, iyA, r0A, r1A, wA, retsp);

  // --- epilogue ---
  k_routing<<<1, 256, 0, stream>>>(cell, rW0, rb0, rW1, rb1, pool, fusedW);
  k_resconv<<<(BB*QQ + 255)/256, 256, 0, stream>>>(retsp, fusedW);
  k_imnet2<<<BB*QQ, 64, 0, stream>>>(cpin, jw[0], jb[0], jw[1], jb[1], jw[2], jb[2], cpred);
  k_final<<<(BB*QQ*3 + 255)/256, 256, 0, stream>>>(retsp, cpred, out);
}
